// MultiModalMambaClassifier2_16724602650792
// MI455X (gfx1250) — matrix operand order repaired and checked by reading, not compiled, not run on hardware
//
#include <hip/hip_runtime.h>

// ---------------------------------------------------------------------------
// MultiModalMambaClassifier (L=1) fully fused for gfx1250 (MI455X, wave32).
//
// Key simplifications (valid because seq len L == 1, zero init state):
//   conv  : h = silu(hs * conv_w[:,3] + conv_b)          (single causal tap)
//   ssm   : y = h * (softplus(dt) * dot(B,C) + D)        (A_log unused)
//
// Pipeline per 64-row batch tile, per modality, per layer (all bf16 WMMA,
// fp32 accumulate, activations resident in LDS):
//   embed -> [rmsnorm -> in_proj -> conv/silu -> x_proj -> B.C -> dt_proj ->
//            ssm collapse * silu(gate) -> out_proj -> +residual] x2
//         -> final rmsnorm -> 0.5*pool -> classifier
// ---------------------------------------------------------------------------

typedef __attribute__((ext_vector_type(16))) __bf16 v16bf;
typedef __attribute__((ext_vector_type(8)))  __bf16 v8bf;
typedef __attribute__((ext_vector_type(8)))  float  v8f;

#define BT      64          // batch rows per workgroup
#define THREADS 128         // 4 waves of 32
#define NBLOCKS (32768 / BT)

// ---- workspace layout (bf16 elements) -------------------------------------
// B-fragments: frag(kt,nt) = 32 lanes x 16 bf16, fragIdx = kt*NT + nt
#define OFF_WM     0                         // methyl_w : K 500->512, N 256 : 16*16*512
#define OFF_WR     131072                    // rna_w    : K 47->64,  N 256 : 2*16*512
#define OFF_BASE   147456
#define L_STRIDE   434176                    // per-layer block
#define OFF_WIN(l) (OFF_BASE + (l)*L_STRIDE)            // in_proj : K 256, N 1024
#define OFF_WXP(l) (OFF_WIN(l) + 262144)                // x_proj  : K 512, N 48
#define OFF_WDT(l) (OFF_WXP(l) + 24576)                 // dt_proj : K 16->32, N 512
#define OFF_WOUT(l) (OFF_WDT(l) + 16384)                // out_proj: K 512, N 256
// total = 1,015,808 bf16 = ~1.94 MB of d_ws

// ---- shared memory carving (307,456 bytes total) ---------------------------
#define SMEM_BYTES 307456

__device__ __forceinline__ unsigned short f2bf(float f) {
    union { float f; unsigned u; } v; v.f = f;
    unsigned r = v.u + 0x7FFFu + ((v.u >> 16) & 1u);   // round-to-nearest-even
    return (unsigned short)(r >> 16);
}
__device__ __forceinline__ float bf2f(unsigned short h) {
    union { float f; unsigned u; } v; v.u = ((unsigned)h) << 16; return v.f;
}
__device__ __forceinline__ float siluf(float x)     { return x / (1.0f + __expf(-x)); }
__device__ __forceinline__ float softplusf(float x) { return (x > 20.0f) ? x : log1pf(__expf(x)); }

__device__ __forceinline__ v16bf cat8(v8bf lo, v8bf hi) {
    return __builtin_shufflevector(lo, hi, 0,1,2,3,4,5,6,7,8,9,10,11,12,13,14,15);
}

// A fragment (16x32 bf16) from row-major LDS. ISA layout: lane m=L%16, hi=L/16;
// element e -> k = 16*(e/8) + 8*hi + (e%8)  => two contiguous 16B runs.
__device__ __forceinline__ v16bf load_a(const unsigned short* base, int ldk,
                                        int row0, int k0, int lane) {
    int mm = lane & 15, hh = lane >> 4;
    const unsigned short* p = base + (row0 + mm) * ldk + k0 + 8 * hh;
    return cat8(*(const v8bf*)p, *(const v8bf*)(p + 16));
}

// B fragment (32x16 bf16), pre-swizzled in d_ws: lane L holds N col L%16, 16 K elems (table order).
__device__ __forceinline__ v16bf load_b(const unsigned short* frags, int fragIdx, int lane) {
    const unsigned short* p = frags + (fragIdx * 32 + lane) * 16;
    return cat8(*(const v8bf*)p, *(const v8bf*)(p + 8));
}

__device__ __forceinline__ v8f wmma_bf16(v16bf a, v16bf b, v8f c) {
    // D(f32 16x16) = A(bf16 16x32) x B(bf16 32x16) + C
    return __builtin_amdgcn_wmma_f32_16x16x32_bf16(false, a, false, b, (short)0, c,
                                                   false, false);
}

// Waves split N (each B fragment loaded once per WG); A reloaded from LDS per tile.
// Epilogue gets (nt, rt, acc); D layout: elem r -> row rt*16 + r + 8*(lane/16),
// col nt*16 + (lane%16).
template<int KT, int NT, typename EPI>
__device__ __forceinline__ void gemm_tiles(const unsigned short* A, int ldk,
                                           const unsigned short* Bfr,
                                           int wave, int lane, EPI epi) {
    for (int nt = wave; nt < NT; nt += 4) {
        v16bf bf[KT];
        #pragma unroll
        for (int kt = 0; kt < KT; ++kt) bf[kt] = load_b(Bfr, kt * NT + nt, lane);
        #pragma unroll
        for (int rt = 0; rt < 4; ++rt) {
            v8f acc = {0.f, 0.f, 0.f, 0.f, 0.f, 0.f, 0.f, 0.f};
            #pragma unroll
            for (int kt = 0; kt < KT; ++kt)
                acc = wmma_bf16(load_a(A, ldk, rt * 16, kt * 32, lane), bf[kt], acc);
            epi(nt, rt, acc);
        }
    }
}

// ---------------------------------------------------------------------------
// Kernel 0: convert fp32 weight [Nout][Kact] -> bf16 WMMA B-fragments (K-padded).
// ---------------------------------------------------------------------------
__global__ void pack_b(const float* __restrict__ W, unsigned short* __restrict__ dst,
                       int Nout, int Kact, int Kpad) {
    int NT = Nout >> 4;
    int total = (Kpad >> 5) * NT * 512;
    for (int idx = blockIdx.x * blockDim.x + threadIdx.x; idx < total;
         idx += gridDim.x * blockDim.x) {
        int e    = idx & 15;
        int L    = (idx >> 4) & 31;
        int frag = idx >> 9;
        int nt = frag % NT;
        int kt = frag / NT;
        int k = kt * 32 + 8 * (L >> 4) + (e & 7) + 16 * (e >> 3);  // elems = K (table order)
        int n = nt * 16 + (L & 15);   // lane%16 = N col
        float v = (k < Kact) ? W[(size_t)n * Kact + k] : 0.f;
        dst[idx] = f2bf(v);
    }
}

// ---------------------------------------------------------------------------
// Kernel 1: fully fused model.
// ---------------------------------------------------------------------------
__global__ void __launch_bounds__(THREADS)
mamba_fused(const float* __restrict__ methyl, const float* __restrict__ rnaf,
            const float* __restrict__ methyl_b, const float* __restrict__ rna_b,
            const float* __restrict__ conv_w, const float* __restrict__ conv_b,
            const float* __restrict__ dt_b, const float* __restrict__ Dw,
            const float* __restrict__ norm_w, const float* __restrict__ norm_f_w,
            const float* __restrict__ cls_w, const float* __restrict__ cls_b,
            const unsigned short* __restrict__ wfr, float* __restrict__ out) {
    extern __shared__ __align__(16) unsigned char smem[];
    float*          lX   = (float*)(smem);                    // [64][256] f32 residual
    float*          lP   = (float*)(smem + 65536);            // [64][256] f32 pooled
    unsigned short* lXn  = (unsigned short*)(smem + 131072);  // [64][256] bf16 rmsnormed
    unsigned short* lH   = (unsigned short*)(smem + 163840);  // [64][512] bf16 h (also input stage)
    unsigned short* lG   = (unsigned short*)(smem + 229376);  // [64][512] bf16 silu(gate) -> y
    unsigned short* lDtr = (unsigned short*)(smem + 294912);  // [64][32]  bf16 dt-rank (K-padded)
    float*          lBm  = (float*)(smem + 299008);           // [64][16]
    float*          lCm  = (float*)(smem + 303104);           // [64][16]
    float*          lSBC = (float*)(smem + 307200);           // [64] dot(B,C)

    const int tid  = threadIdx.x;
    const int lane = tid & 31;
    const int wave = tid >> 5;
    const int nn   = lane & 15;   // D-tile column within tile
    const int hh   = lane >> 4;   // D-tile row-half select
    const int gr0  = blockIdx.x * BT;

    for (int mod = 0; mod < 2; ++mod) {
        // ---- stage input tile -> bf16 in lH (zero K-padded) ----
        if (mod == 0) {
            for (int idx = tid; idx < BT * 512; idx += THREADS) {
                int r = idx >> 9, c = idx & 511;
                lH[idx] = f2bf(c < 500 ? methyl[(size_t)(gr0 + r) * 500 + c] : 0.f);
            }
        } else {
            for (int idx = tid; idx < BT * 64; idx += THREADS) {
                int r = idx >> 6, c = idx & 63;
                lH[idx] = f2bf(c < 47 ? rnaf[(size_t)(gr0 + r) * 47 + c] : 0.f);
            }
        }
        __syncthreads();

        // ---- embedding GEMM -> lX ----
        if (mod == 0) {
            gemm_tiles<16, 16>(lH, 512, wfr + OFF_WM, wave, lane,
                [&](int nt, int rt, v8f acc) {
                    int col = nt * 16 + nn;
                    float b = methyl_b[col];
                    #pragma unroll
                    for (int r = 0; r < 8; ++r)
                        lX[(rt * 16 + r + 8 * hh) * 256 + col] = acc[r] + b;
                });
        } else {
            gemm_tiles<2, 16>(lH, 64, wfr + OFF_WR, wave, lane,
                [&](int nt, int rt, v8f acc) {
                    int col = nt * 16 + nn;
                    float b = rna_b[col];
                    #pragma unroll
                    for (int r = 0; r < 8; ++r)
                        lX[(rt * 16 + r + 8 * hh) * 256 + col] = acc[r] + b;
                });
        }
        __syncthreads();

        for (int l = 0; l < 2; ++l) {
            // ---- rmsnorm(x) * norm_w -> lXn (bf16); zero dt K-pad ----
            if (tid < BT) {
                int row = tid;
                float ss = 0.f;
                for (int c = 0; c < 256; ++c) { float v = lX[row * 256 + c]; ss += v * v; }
                float inv = rsqrtf(ss * (1.f / 256.f) + 1e-5f);
                for (int c = 0; c < 256; ++c)
                    lXn[row * 256 + c] = f2bf(lX[row * 256 + c] * inv * norm_w[l * 256 + c]);
                for (int c = 16; c < 32; ++c) lDtr[row * 32 + c] = 0;
            }
            __syncthreads();

            // ---- in_proj; epilogue: conv tap3 + silu -> lH, silu(gate) -> lG ----
            gemm_tiles<8, 64>(lXn, 256, wfr + OFF_WIN(l), wave, lane,
                [&](int nt, int rt, v8f acc) {
                    int col = nt * 16 + nn;
                    if (col < 512) {
                        float c3 = conv_w[((size_t)l * 512 + col) * 4 + 3];
                        float cb = conv_b[l * 512 + col];
                        #pragma unroll
                        for (int r = 0; r < 8; ++r) {
                            int row = rt * 16 + r + 8 * hh;
                            lH[row * 512 + col] = f2bf(siluf(acc[r] * c3 + cb));
                        }
                    } else {
                        int gc = col - 512;
                        #pragma unroll
                        for (int r = 0; r < 8; ++r) {
                            int row = rt * 16 + r + 8 * hh;
                            lG[row * 512 + gc] = f2bf(siluf(acc[r]));
                        }
                    }
                });
            __syncthreads();

            // ---- x_proj -> dt-rank (bf16), B, C ----
            gemm_tiles<16, 3>(lH, 512, wfr + OFF_WXP(l), wave, lane,
                [&](int nt, int rt, v8f acc) {
                    #pragma unroll
                    for (int r = 0; r < 8; ++r) {
                        int row = rt * 16 + r + 8 * hh;
                        if (nt == 0)      lDtr[row * 32 + nn] = f2bf(acc[r]);
                        else if (nt == 1) lBm[row * 16 + nn] = acc[r];
                        else              lCm[row * 16 + nn] = acc[r];
                    }
                });
            __syncthreads();

            if (tid < BT) {
                float s = 0.f;
                for (int n2 = 0; n2 < 16; ++n2) s += lBm[tid * 16 + n2] * lCm[tid * 16 + n2];
                lSBC[tid] = s;
            }
            __syncthreads();

            // ---- dt_proj; epilogue: softplus, SSM collapse, * silu(gate) -> y in lG ----
            gemm_tiles<1, 32>(lDtr, 32, wfr + OFF_WDT(l), wave, lane,
                [&](int nt, int rt, v8f acc) {
                    int col = nt * 16 + nn;
                    float db = dt_b[l * 512 + col];
                    float dd = Dw[l * 512 + col];
                    #pragma unroll
                    for (int r = 0; r < 8; ++r) {
                        int row = rt * 16 + r + 8 * hh;
                        float dt = softplusf(acc[r] + db);
                        float h  = bf2f(lH[row * 512 + col]);
                        float g  = bf2f(lG[row * 512 + col]);
                        lG[row * 512 + col] = f2bf(h * (dt * lSBC[row] + dd) * g);
                    }
                });
            __syncthreads();

            // ---- out_proj + residual into lX ----
            gemm_tiles<16, 16>(lG, 512, wfr + OFF_WOUT(l), wave, lane,
                [&](int nt, int rt, v8f acc) {
                    int col = nt * 16 + nn;
                    #pragma unroll
                    for (int r = 0; r < 8; ++r) {
                        int row = rt * 16 + r + 8 * hh;
                        lX[row * 256 + col] += acc[r];
                    }
                });
            __syncthreads();
        }

        // ---- final rmsnorm, accumulate 0.5 * into pooled ----
        if (tid < BT) {
            int row = tid;
            float ss = 0.f;
            for (int c = 0; c < 256; ++c) { float v = lX[row * 256 + c]; ss += v * v; }
            float inv = rsqrtf(ss * (1.f / 256.f) + 1e-5f);
            for (int c = 0; c < 256; ++c) {
                float v = lX[row * 256 + c] * inv * norm_f_w[c] * 0.5f;
                if (mod == 0) lP[row * 256 + c] = v;
                else          lP[row * 256 + c] += v;
            }
        }
        __syncthreads();
    }

    // ---- classifier: [64,256] x [5,256]^T + bias (tiny; scalar fp32) ----
    if (tid < BT) {
        int row = tid;
        size_t gr = (size_t)gr0 + row;
        for (int c2 = 0; c2 < 5; ++c2) {
            float a = cls_b[c2];
            for (int k = 0; k < 256; ++k) a += lP[row * 256 + k] * cls_w[c2 * 256 + k];
            out[gr * 5 + c2] = a;
        }
    }
}

// ---------------------------------------------------------------------------
extern "C" void kernel_launch(void* const* d_in, const int* in_sizes, int n_in,
                              void* d_out, int out_size, void* d_ws, size_t ws_size,
                              hipStream_t stream) {
    (void)in_sizes; (void)n_in; (void)out_size; (void)ws_size;

    const float* methyl     = (const float*)d_in[0];
    const float* rnaf       = (const float*)d_in[1];
    const float* methyl_w   = (const float*)d_in[2];
    const float* methyl_b   = (const float*)d_in[3];
    const float* rna_w      = (const float*)d_in[4];
    const float* rna_b      = (const float*)d_in[5];
    const float* in_proj_w  = (const float*)d_in[6];
    const float* conv_w     = (const float*)d_in[7];
    const float* conv_b     = (const float*)d_in[8];
    const float* x_proj_w   = (const float*)d_in[9];
    const float* dt_proj_w  = (const float*)d_in[10];
    const float* dt_proj_b  = (const float*)d_in[11];
    // d_in[12] = A_log: provably unused when L == 1 (state starts at zero)
    const float* Dp         = (const float*)d_in[13];
    const float* out_proj_w = (const float*)d_in[14];
    const float* norm_w     = (const float*)d_in[15];
    const float* norm_f_w   = (const float*)d_in[16];
    const float* cls_w      = (const float*)d_in[17];
    const float* cls_b      = (const float*)d_in[18];

    unsigned short* ws = (unsigned short*)d_ws;

    auto pk = [&](const float* W, int off, int Nout, int Kact, int Kpad) {
        int total  = (Kpad / 32) * (Nout / 16) * 512;
        int blocks = (total + 255) / 256;
        pack_b<<<blocks, 256, 0, stream>>>(W, ws + off, Nout, Kact, Kpad);
    };

    pk(methyl_w, OFF_WM, 256, 500, 512);
    pk(rna_w,    OFF_WR, 256, 47, 64);
    for (int l = 0; l < 2; ++l) {
        pk(in_proj_w  + (size_t)l * 1024 * 256, OFF_WIN(l),  1024, 256, 256);
        pk(x_proj_w   + (size_t)l * 48 * 512,   OFF_WXP(l),  48,   512, 512);
        pk(dt_proj_w  + (size_t)l * 512 * 16,   OFF_WDT(l),  512,  16,  32);
        pk(out_proj_w + (size_t)l * 256 * 512,  OFF_WOUT(l), 256,  512, 512);
    }

    hipFuncSetAttribute((const void*)mamba_fused,
                        hipFuncAttributeMaxDynamicSharedMemorySize, SMEM_BYTES);
    mamba_fused<<<NBLOCKS, THREADS, SMEM_BYTES, stream>>>(
        methyl, rnaf, methyl_b, rna_b, conv_w, conv_b, dt_proj_b, Dp,
        norm_w, norm_f_w, cls_w, cls_b, (const unsigned short*)ws, (float*)d_out);
}
